// VQVAEModel_56049323213430
// MI455X (gfx1250) — compile-verified, matrix operand order checked by reading, not run on hardware
//
#include <hip/hip_runtime.h>

typedef __attribute__((ext_vector_type(16))) __bf16 v16bf;
typedef __attribute__((ext_vector_type(8)))  __bf16 v8bf;
typedef __attribute__((ext_vector_type(8)))  float  v8f;

#define NROWS      65536          // B*S = 16*4096
#define DIM        256            // D
#define KCODES     4096           // K
#define TILE_COLS  32             // codebook columns per LDS tile
#define NTILES     (KCODES / TILE_COLS)
#define LDS_STRIDE 264            // 256 + 8 bf16 pad -> conflict-free b128 reads
#define BUF_ELEMS  (TILE_COLS * LDS_STRIDE)
#define ROWS_PER_WAVE 32          // two 16-row A tiles -> each B fragment feeds 2 WMMAs
#define ROWS_PER_BLOCK 256        // 8 waves
#define LOSS_SCALE (0.25f / 16777216.0f)   // beta / (N*D)

__device__ __forceinline__ float sqdiff4(const float4& x, const float4& q) {
    float a = x.x - q.x, b = x.y - q.y, c = x.z - q.z, d = x.w - q.w;
    return a * a + b * b + c * c + d * d;
}

__device__ __forceinline__ v16bf catbf(v8bf lo, v8bf hi) {
    v16bf r;
#pragma unroll
    for (int i = 0; i < 8; ++i) { r[i] = lo[i]; r[8 + i] = hi[i]; }
    return r;
}

__device__ __forceinline__ void load_afrags(const float* __restrict__ base,
                                            int h, v16bf* afrag) {
#pragma unroll
    for (int kc = 0; kc < 8; ++kc) {
        const float* p0 = base + kc * 32 + 8 * h;     // K = k0+8h .. +7, then +16
        float4 f0 = ((const float4*)p0)[0];
        float4 f1 = ((const float4*)p0)[1];
        float4 f2 = ((const float4*)p0)[4];
        float4 f3 = ((const float4*)p0)[5];
        afrag[kc][0]  = (__bf16)f0.x; afrag[kc][1]  = (__bf16)f0.y;
        afrag[kc][2]  = (__bf16)f0.z; afrag[kc][3]  = (__bf16)f0.w;
        afrag[kc][4]  = (__bf16)f1.x; afrag[kc][5]  = (__bf16)f1.y;
        afrag[kc][6]  = (__bf16)f1.z; afrag[kc][7]  = (__bf16)f1.w;
        afrag[kc][8]  = (__bf16)f2.x; afrag[kc][9]  = (__bf16)f2.y;
        afrag[kc][10] = (__bf16)f2.z; afrag[kc][11] = (__bf16)f2.w;
        afrag[kc][12] = (__bf16)f3.x; afrag[kc][13] = (__bf16)f3.y;
        afrag[kc][14] = (__bf16)f3.z; afrag[kc][15] = (__bf16)f3.w;
    }
}

// Kernel 1: per-code squared norms (fp32), bf16 codebook copy, zero loss accumulator.
__global__ __launch_bounds__(256) void vq_prep_kernel(const float* __restrict__ cb,
                                                      float* __restrict__ nrm,
                                                      __bf16* __restrict__ cb16,
                                                      float* __restrict__ lossPtr) {
    if (blockIdx.x == 0 && threadIdx.x == 0) *lossPtr = 0.0f;
    const int wave = threadIdx.x >> 5;
    const int lane = threadIdx.x & 31;
    const int row  = blockIdx.x * 8 + wave;
    const float4* p = (const float4*)(cb + (size_t)row * DIM) + lane * 2;
    float4 a = p[0], b = p[1];
    // bf16 copy of this row slice
    v8bf v;
    v[0] = (__bf16)a.x; v[1] = (__bf16)a.y; v[2] = (__bf16)a.z; v[3] = (__bf16)a.w;
    v[4] = (__bf16)b.x; v[5] = (__bf16)b.y; v[6] = (__bf16)b.z; v[7] = (__bf16)b.w;
    *(v8bf*)(cb16 + (size_t)row * DIM + lane * 8) = v;
    // squared norm
    float s = a.x * a.x + a.y * a.y + a.z * a.z + a.w * a.w +
              b.x * b.x + b.y * b.y + b.z * b.z + b.w * b.w;
#pragma unroll
    for (int m = 1; m < 32; m <<= 1) s += __shfl_xor(s, m, 32);
    if (lane == 0) nrm[row] = s;
}

// Each thread async-copies its 64-byte slice of one 32x256 bf16 tile: global -> LDS.
__device__ __forceinline__ void issue_tile_copy(const __bf16* __restrict__ cb16,
                                                int tileIdx, __bf16* bufPtr,
                                                int tcol, int tseg) {
    unsigned lds = (unsigned)(size_t)(bufPtr + tcol * LDS_STRIDE + tseg * 32);
    unsigned long long g =
        (unsigned long long)(size_t)(cb16 + (size_t)(tileIdx * TILE_COLS + tcol) * DIM + tseg * 32);
    asm volatile("global_load_async_to_lds_b128 %0, %1, off offset:0"  :: "v"(lds), "v"(g) : "memory");
    asm volatile("global_load_async_to_lds_b128 %0, %1, off offset:16" :: "v"(lds), "v"(g) : "memory");
    asm volatile("global_load_async_to_lds_b128 %0, %1, off offset:32" :: "v"(lds), "v"(g) : "memory");
    asm volatile("global_load_async_to_lds_b128 %0, %1, off offset:48" :: "v"(lds), "v"(g) : "memory");
}

// Kernel 2: WMMA distance GEMM + argmin + gather + commitment loss.
__global__ __launch_bounds__(256) void vq_main_kernel(const float* __restrict__ inp,
                                                      const float* __restrict__ cb,
                                                      const float* __restrict__ nrm,
                                                      const __bf16* __restrict__ cb16,
                                                      float* __restrict__ outQ,
                                                      float* __restrict__ lossPtr,
                                                      float* __restrict__ outIdx) {
    __shared__ __align__(16) __bf16 tileB[2 * BUF_ELEMS];   // ~33 KB double buffer

    const int wave    = threadIdx.x >> 5;
    const int lane    = threadIdx.x & 31;
    const int rowBase = blockIdx.x * ROWS_PER_BLOCK + wave * ROWS_PER_WAVE;
    const int mrow    = lane & 15;
    const int h       = lane >> 4;

    // ---- Two 16x256 A tiles per wave, kept as bf16 WMMA fragments ----
    v16bf afragA[8], afragB[8];
    load_afrags(inp + (size_t)(rowBase + mrow) * DIM,      h, afragA);
    load_afrags(inp + (size_t)(rowBase + 16 + mrow) * DIM, h, afragB);

    float minv[2][8];
    int   mini[2][8];
#pragma unroll
    for (int rt = 0; rt < 2; ++rt)
#pragma unroll
        for (int j = 0; j < 8; ++j) { minv[rt][j] = 3.4e38f; mini[rt][j] = 0; }

    const int tcol = threadIdx.x >> 3;   // 0..31: tile column copied by this thread
    const int tseg = threadIdx.x & 7;    // 0..7 : 64-byte segment within the column

    // ---- prologue: async-copy tile 0 into LDS buffer 0 ----
    issue_tile_copy(cb16, 0, tileB, tcol, tseg);

    for (int t = 0; t < NTILES; ++t) {
        const int colBase = t * TILE_COLS;
        __bf16* buf = tileB + (t & 1) * BUF_ELEMS;

        asm volatile("s_wait_asynccnt 0" ::: "memory");  // my slice of tile t landed
        __syncthreads();                                  // everyone's slice landed

        // ---- issue async copies for the next tile (other buffer) ----
        if (t + 1 < NTILES)
            issue_tile_copy(cb16, t + 1, tileB + ((t + 1) & 1) * BUF_ELEMS, tcol, tseg);

        // ---- per 16-col sub-tile: one B fragment feeds two row-tile WMMAs ----
#pragma unroll
        for (int sub = 0; sub < 2; ++sub) {
            const float nn = nrm[colBase + sub * 16 + mrow];
            const __bf16* bp = buf + (sub * 16 + mrow) * LDS_STRIDE + 8 * h;

            v8bf lo = *(const v8bf*)(bp);
            v8bf hi = *(const v8bf*)(bp + 16);
            v8f acc0 = {0.f, 0.f, 0.f, 0.f, 0.f, 0.f, 0.f, 0.f};
            v8f acc1 = {0.f, 0.f, 0.f, 0.f, 0.f, 0.f, 0.f, 0.f};
#pragma unroll
            for (int kc = 0; kc < 8; ++kc) {
                v16bf b = catbf(lo, hi);
                if (kc < 7) {                     // prefetch next fragment
                    lo = *(const v8bf*)(bp + (kc + 1) * 32);
                    hi = *(const v8bf*)(bp + (kc + 1) * 32 + 16);
                }
                acc0 = __builtin_amdgcn_wmma_f32_16x16x32_bf16(
                    false, afragA[kc], false, b, (short)0, acc0, false, false);
                acc1 = __builtin_amdgcn_wmma_f32_16x16x32_bf16(
                    false, afragB[kc], false, b, (short)0, acc1, false, false);
            }

            const int col = colBase + sub * 16 + mrow;
#pragma unroll
            for (int j = 0; j < 8; ++j) {
                float s0 = nn - 2.0f * acc0[j];   // ||e||^2 - 2 x.e
                if (s0 < minv[0][j]) { minv[0][j] = s0; mini[0][j] = col; }
                float s1 = nn - 2.0f * acc1[j];
                if (s1 < minv[1][j]) { minv[1][j] = s1; mini[1][j] = col; }
            }
        }
    }

    // ---- cross-lane (min, argmin) reduction within each 16-lane half ----
#pragma unroll
    for (int rt = 0; rt < 2; ++rt) {
#pragma unroll
        for (int j = 0; j < 8; ++j) {
#pragma unroll
            for (int off = 1; off < 16; off <<= 1) {
                float ov = __shfl_xor(minv[rt][j], off, 32);
                int   oi = __shfl_xor(mini[rt][j], off, 32);
                if (ov < minv[rt][j] || (ov == minv[rt][j] && oi < mini[rt][j])) {
                    minv[rt][j] = ov; mini[rt][j] = oi;
                }
            }
        }
        // lane 0 holds rows +0..7, lane 16 holds rows +8..15 of this row tile
        if ((lane & 15) == 0) {
#pragma unroll
            for (int j = 0; j < 8; ++j)
                outIdx[rowBase + rt * 16 + h * 8 + j] = (float)mini[rt][j];
        }
    }

    // ---- gather winning codes to output + commitment loss partial ----
    float lsum = 0.0f;
#pragma unroll
    for (int rt = 0; rt < 2; ++rt) {
#pragma unroll
        for (int r = 0; r < 16; ++r) {
            const int idx = __shfl(mini[rt][r & 7], (r >> 3) << 4, 32);
            const int row = rowBase + rt * 16 + r;
            const float4* cbp = (const float4*)(cb  + (size_t)idx * DIM) + lane * 2;
            const float4* xp  = (const float4*)(inp + (size_t)row * DIM) + lane * 2;
            float4*       op  = (float4*)(outQ + (size_t)row * DIM) + lane * 2;
            float4 q0 = cbp[0], q1 = cbp[1];
            float4 x0 = xp[0],  x1 = xp[1];
            op[0] = q0; op[1] = q1;
            lsum += sqdiff4(x0, q0) + sqdiff4(x1, q1);
        }
    }
#pragma unroll
    for (int m = 1; m < 32; m <<= 1) lsum += __shfl_xor(lsum, m, 32);
    if (lane == 0) atomicAdd(lossPtr, lsum * LOSS_SCALE);
}

extern "C" void kernel_launch(void* const* d_in, const int* in_sizes, int n_in,
                              void* d_out, int out_size, void* d_ws, size_t ws_size,
                              hipStream_t stream) {
    const float* inp = (const float*)d_in[0];   // (16,4096,256) f32
    const float* cb  = (const float*)d_in[1];   // (4096,256) f32
    float* outQ    = (float*)d_out;                          // quantized_st: N*D
    float* lossPtr = outQ + (size_t)NROWS * DIM;             // commitment_loss: 1
    float* outIdx  = lossPtr + 1;                            // encoding_indices: N
    float*  nrm  = (float*)d_ws;                             // K fp32 norms
    __bf16* cb16 = (__bf16*)(nrm + KCODES);                  // K*D bf16 codebook (2 MB)

    vq_prep_kernel<<<KCODES / 8, 256, 0, stream>>>(cb, nrm, cb16, lossPtr);
    vq_main_kernel<<<NROWS / ROWS_PER_BLOCK, 256, 0, stream>>>(inp, cb, nrm, cb16,
                                                               outQ, lossPtr, outIdx);
}
